// SPECN_42734924595751
// MI455X (gfx1250) — matrix operand order repaired and checked by reading, not compiled, not run on hardware
//
#include <hip/hip_runtime.h>
#include <hip/hip_bf16.h>

#define B_TOT 4096
#define L_SEQ 5
#define T_ITM 3
#define D_DIM 128
#define NHc 16
#define NVc 4
#define NUc 8
#define NU1c 4
#define W2DIM 276   // 9*NH + D + NU1
#define BPB 16      // batch rows per block
#define NTHREADS 512

typedef __bf16 v16bf __attribute__((ext_vector_type(16)));
typedef float  v8f   __attribute__((ext_vector_type(8)));

__device__ __forceinline__ float sigmoidf(float x) { return 1.0f / (1.0f + __expf(-x)); }

// full-wave (wave32) butterfly sum; result replicated in all lanes
__device__ __forceinline__ float wred(float v) {
  v += __shfl_xor(v, 16, 32);
  v += __shfl_xor(v,  8, 32);
  v += __shfl_xor(v,  4, 32);
  v += __shfl_xor(v,  2, 32);
  v += __shfl_xor(v,  1, 32);
  return v;
}

__global__ __launch_bounds__(NTHREADS, 1)
void caser_fused_kernel(
    const int*   __restrict__ seq_var,   const int*   __restrict__ user_var,
    const int*   __restrict__ item_var,
    const float* __restrict__ item_emb_w, const float* __restrict__ user_emb_w,
    const float* __restrict__ conv_v_w,   const float* __restrict__ conv_v_b,
    const float* __restrict__ conv_h_w,   const float* __restrict__ conv_h_b,
    const float* __restrict__ conv_u_w,   const float* __restrict__ conv_u_b,
    const float* __restrict__ conv_u1_w,  const float* __restrict__ conv_u1_b,
    const float* __restrict__ W2_w,       const float* __restrict__ b2_w,
    const float* __restrict__ CW1,        const float* __restrict__ CW2,
    const float* __restrict__ CW00,       const float* __restrict__ CW11,
    float* __restrict__ out)
{
  // region: phases 1-3 => item staging [16][5][128] f32 (40KB)
  //         phases 4-7 => sh_A [16][512] bf16 (16KB) + CW1 [16][128] f32 (8KB)
  __shared__ __align__(16) char s_region[40960];
  __shared__ float sh_x[BPB][W2DIM];   // per-b feature vector x (17.25KB)

  const int tid  = threadIdx.x;
  const int lane = tid & 31;
  const int wv   = tid >> 5;          // 0..15, one wave per local batch row
  const int b0   = blockIdx.x * BPB;
  const int bl   = wv;
  const int bg   = b0 + bl;

  float* sh_items = (float*)s_region;  // [16*5][128]

  // ---- Prefetch W2 gather rows for the final phase (global_prefetch_b8) ----
  #pragma unroll
  for (int t = 0; t < T_ITM; ++t) {
    int idx = item_var[bg * T_ITM + t];
    if (lane < 9)
      __builtin_prefetch(W2_w + (size_t)idx * W2DIM + lane * 32, 0, 0);
  }

  // ---- Phase 1: cooperative gather of item embeddings into LDS ----
  for (int v = tid; v < BPB * L_SEQ * 32; v += NTHREADS) {
    int row = v >> 5;                  // bl*5 + s
    int d4  = v & 31;
    int bb  = row / L_SEQ, s = row - bb * L_SEQ;
    int idx = seq_var[(b0 + bb) * L_SEQ + s];
    ((float4*)(sh_items + row * D_DIM))[d4] =
        ((const float4*)(item_emb_w + (size_t)idx * D_DIM))[d4];
  }
  __syncthreads();

  // per-lane item values: d = lane + 32*q
  float it[L_SEQ][4];
  #pragma unroll
  for (int s = 0; s < L_SEQ; ++s)
    #pragma unroll
    for (int q = 0; q < 4; ++q)
      it[s][q] = sh_items[(bl * L_SEQ + s) * D_DIM + lane + 32 * q];

  // ---- Phase 2: vertical conv + sigmoid; only xsumV needed by routing ----
  float xv[4];
  {
    float eff[NVc][L_SEQ], bv[NVc];
    #pragma unroll
    for (int o = 0; o < NVc; ++o) {
      bv[o] = conv_v_b[o];
      #pragma unroll
      for (int s = 0; s < L_SEQ; ++s)
        eff[o][s] = conv_v_w[o * 10 + 2 * s] + conv_v_w[o * 10 + 2 * s + 1];
    }
    #pragma unroll
    for (int q = 0; q < 4; ++q) {
      float x = 0.0f;
      #pragma unroll
      for (int o = 0; o < NVc; ++o) {
        float a = bv[o];
        #pragma unroll
        for (int s = 0; s < L_SEQ; ++s) a += it[s][q] * eff[o][s];
        x += sigmoidf(a);
      }
      xv[q] = x;
    }
  }

  // ---- Phase 3: horizontal conv + sigmoid; only xsumH needed by routing ----
  float xh[NHc];
  for (int o = 0; o < NHc; ++o) {
    float pa0[L_SEQ], pa1[L_SEQ];
    #pragma unroll
    for (int s = 0; s < L_SEQ; ++s) { pa0[s] = 0.0f; pa1[s] = 0.0f; }
    #pragma unroll
    for (int q = 0; q < 4; ++q) {
      int d = lane + 32 * q;
      float w0 = conv_h_w[(o * 2 + 0) * D_DIM + d];
      float w1 = conv_h_w[(o * 2 + 1) * D_DIM + d];
      #pragma unroll
      for (int s = 0; s < L_SEQ; ++s) {
        pa0[s] += it[s][q] * w0;
        pa1[s] += it[s][q] * w1;
      }
    }
    float a0[L_SEQ], a1[L_SEQ];
    #pragma unroll
    for (int s = 0; s < L_SEQ; ++s) { a0[s] = wred(pa0[s]); a1[s] = wred(pa1[s]); }
    float bh = conv_h_b[o];
    float x = 0.0f;
    #pragma unroll
    for (int t = 0; t < 9; ++t)
      x += sigmoidf(a0[t >> 1] + a1[(t + 1) >> 1] + bh);  // merged[t]=item[t/2]
    xh[o] = x;
  }
  __syncthreads();  // items region now free for reuse

  __bf16* sh_A   = (__bf16*)s_region;            // [16][512] bf16 A-matrix
  float*  sh_CW1 = (float*)(s_region + 16384);   // [16][128]

  // stage CW1 into LDS (needed by routing H)
  for (int v = tid; v < NHc * D_DIM; v += NTHREADS) sh_CW1[v] = CW1[v];

  // ---- Phase 4: routing V (4 capsules, I=128, J=4); hat = xsumV[i]*CW2[i,j] ----
  {
    float cw2[4][4];
    #pragma unroll
    for (int q = 0; q < 4; ++q)
      #pragma unroll
      for (int j = 0; j < 4; ++j)
        cw2[q][j] = CW2[(lane + 32 * q) * 4 + j];
    float blog[4][4];
    #pragma unroll
    for (int c = 0; c < 4; ++c)
      #pragma unroll
      for (int j = 0; j < 4; ++j) blog[c][j] = 0.0f;

    for (int r = 0; r < 3; ++r) {
      float cs[4][4];
      #pragma unroll
      for (int j = 0; j < 4; ++j) {        // softmax over capsule axis
        float m = blog[0][j];
        #pragma unroll
        for (int c = 1; c < 4; ++c) m = fmaxf(m, blog[c][j]);
        float sum = 0.0f;
        #pragma unroll
        for (int c = 0; c < 4; ++c) { float e = __expf(blog[c][j] - m); cs[c][j] = e; sum += e; }
        float inv = 1.0f / sum;
        #pragma unroll
        for (int c = 0; c < 4; ++c) cs[c][j] *= inv;
      }
      #pragma unroll
      for (int c = 0; c < 4; ++c) {
        float oq[4]; float sp = 0.0f;
        #pragma unroll
        for (int q = 0; q < 4; ++q) {
          float t = 0.0f;
          #pragma unroll
          for (int j = 0; j < 4; ++j) t += cw2[q][j] * cs[c][j];
          oq[q] = xv[q] * t;
          sp += oq[q] * oq[q];
        }
        float sq = wred(sp);
        float scale = sq / ((1.0f + sq) * sqrtf(sq) + 1e-30f);  // squash scale
        if (r < 2) {
          #pragma unroll
          for (int j = 0; j < 4; ++j) {
            float p = 0.0f;
            #pragma unroll
            for (int q = 0; q < 4; ++q) p += xv[q] * cw2[q][j] * oq[q];
            blog[c][j] += scale * wred(p);
          }
        } else {  // final routed out_v -> bf16 A-matrix row, k = c*128 + i
          #pragma unroll
          for (int q = 0; q < 4; ++q)
            sh_A[bl * 512 + c * D_DIM + lane + 32 * q] = (__bf16)(scale * oq[q]);
        }
      }
    }
  }
  __syncthreads();  // CW1 visible to all waves

  // ---- Phase 5: routing H (9 capsules, I=16, J=128); writes C into sh_x[:,0:144] ----
  {
    float blog[9][4];   // b-logits for this lane's 4 j values (j = lane+32q)
    #pragma unroll
    for (int c = 0; c < 9; ++c)
      #pragma unroll
      for (int q = 0; q < 4; ++q) blog[c][q] = 0.0f;

    for (int r = 0; r < 3; ++r) {
      float cs[9][4];
      #pragma unroll
      for (int q = 0; q < 4; ++q) {       // softmax over 9 capsules per j
        float m = blog[0][q];
        #pragma unroll
        for (int c = 1; c < 9; ++c) m = fmaxf(m, blog[c][q]);
        float sum = 0.0f;
        #pragma unroll
        for (int c = 0; c < 9; ++c) { float e = __expf(blog[c][q] - m); cs[c][q] = e; sum += e; }
        float inv = 1.0f / sum;
        #pragma unroll
        for (int c = 0; c < 9; ++c) cs[c][q] *= inv;
      }
      for (int c = 0; c < 9; ++c) {
        float u[NHc];
        #pragma unroll
        for (int i = 0; i < NHc; ++i) {   // u[c,i] = sum_j CW1[i,j]*cs[c,j]
          float p = 0.0f;
          #pragma unroll
          for (int q = 0; q < 4; ++q) p += sh_CW1[i * D_DIM + lane + 32 * q] * cs[c][q];
          u[i] = wred(p);
        }
        float sq = 0.0f;                  // replicated across lanes
        #pragma unroll
        for (int i = 0; i < NHc; ++i) { float o = xh[i] * u[i]; sq += o * o; }
        float scale = sq / ((1.0f + sq) * sqrtf(sq) + 1e-30f);
        if (r < 2) {
          #pragma unroll
          for (int q = 0; q < 4; ++q) {   // b[c,j] += sum_i xh[i]^2*u[i]*CW1[i,j]*scale
            float dl = 0.0f;
            #pragma unroll
            for (int i = 0; i < NHc; ++i)
              dl += xh[i] * xh[i] * u[i] * sh_CW1[i * D_DIM + lane + 32 * q];
            blog[c][q] += scale * dl;
          }
        } else if (lane == 0) {           // routed out_h -> GEMM C operand
          #pragma unroll
          for (int i = 0; i < NHc; ++i)
            sh_x[bl][c * NHc + i] = scale * xh[i] * u[i];
        }
      }
    }
  }

  // ---- Phase 6: user branch: x[144:272]=user_emb, x[272:276]=user_merge0 ----
  {
    int uidx = user_var[bg];
    float e[4];
    #pragma unroll
    for (int q = 0; q < 4; ++q) {
      e[q] = user_emb_w[(size_t)uidx * D_DIM + lane + 32 * q];
      sh_x[bl][144 + lane + 32 * q] = e[q];
    }
    float cu_w[NUc], cu_b[NUc];
    #pragma unroll
    for (int u = 0; u < NUc; ++u) { cu_w[u] = conv_u_w[u]; cu_b[u] = conv_u_b[u]; }
    float acc[NU1c] = {0, 0, 0, 0}, du1[NU1c] = {0, 0, 0, 0};
    #pragma unroll
    for (int q = 0; q < 4; ++q) {
      int d = lane + 32 * q;
      float ed = e[q];
      #pragma unroll
      for (int u = 0; u < NUc; ++u) {
        float s = sigmoidf(ed * cu_w[u] + cu_b[u]);
        const float* row = CW11 + (size_t)(u * D_DIM + d) * NU1c;
        #pragma unroll
        for (int j = 0; j < NU1c; ++j) acc[j] += s * row[j];
      }
      #pragma unroll
      for (int j = 0; j < NU1c; ++j) du1[j] += ed * conv_u1_w[j * D_DIM + d];
    }
    #pragma unroll
    for (int j = 0; j < NU1c; ++j) {
      float a  = wred(acc[j]);
      float d1 = wred(du1[j]);
      if (lane == 0)
        sh_x[bl][272 + j] = fmaxf(a + sigmoidf(d1 + conv_u1_b[j]), 0.0f);
    }
  }
  __syncthreads();  // sh_A complete + sh_x C-columns complete

  // ---- Phase 7: WMMA GEMM: sh_x[:,0:144] = relu(A(16x512) * CW00(512x144) + C) ----
  // bf16 v_wmma_f32_16x16x32_bf16, 9 N-tiles (one per wave 0..8), 16 K-steps,
  // fully unrolled so the scheduler can pipeline B-loads/converts across the
  // WMMA chain and issue the matrix ops back-to-back.
  if (wv < 9) {
    const int n0 = wv * 16;
    const int mA = lane & 15;
    const int g  = lane >> 4;
    const int nC = n0 + (lane & 15);
    v8f acc;
    #pragma unroll
    for (int r = 0; r < 8; ++r) acc[r] = sh_x[r + 8 * g][nC];  // C/D layout: M = r + 8*g
    #pragma unroll
    for (int kt = 0; kt < 16; ++kt) {
      const int k0 = kt * 32;
      v16bf af, bfr;
      // A 16x32 bf16 layout: regs 0-3 K = g*8 + [0..7], regs 4-7 K = 16 + g*8 + [0..7]
      #pragma unroll
      for (int rr = 0; rr < 4; ++rr) {
        int kA = k0 + g * 8 + 2 * rr;
        af[2 * rr]     = sh_A[mA * 512 + kA];
        af[2 * rr + 1] = sh_A[mA * 512 + kA + 1];
      }
      #pragma unroll
      for (int rr = 4; rr < 8; ++rr) {
        int kA = k0 + 16 + g * 8 + 2 * (rr - 4);
        af[2 * rr]     = sh_A[mA * 512 + kA];
        af[2 * rr + 1] = sh_A[mA * 512 + kA + 1];
      }
      // B 32x16 bf16 layout: regs 0-3 K = g*8 + [0..7], regs 4-7 K = 16 + g*8 + [0..7] (2 per reg)
      #pragma unroll
      for (int rr = 0; rr < 8; ++rr) {
        int kB = k0 + g * 8 + 2 * rr + (rr >= 4 ? 8 : 0);
        bfr[2 * rr]     = (__bf16)CW00[kB * 144 + nC];        // L2-resident
        bfr[2 * rr + 1] = (__bf16)CW00[(kB + 1) * 144 + nC];
      }
      acc = __builtin_amdgcn_wmma_f32_16x16x32_bf16(
          /*neg_a=*/false, af, /*neg_b=*/false, bfr,
          /*c_mod=*/(short)0, acc, /*reuse_a=*/false, /*reuse_b=*/false);
    }
    #pragma unroll
    for (int r = 0; r < 8; ++r) sh_x[r + 8 * g][nC] = fmaxf(acc[r], 0.0f);
  }
  __syncthreads();

  // ---- Phase 8: res[b,t] = W2_w[item_var[b,t]] . x[b] + b2 ----
  #pragma unroll
  for (int t = 0; t < T_ITM; ++t) {
    int idx = item_var[bg * T_ITM + t];
    const float* w2 = W2_w + (size_t)idx * W2DIM;
    float p = 0.0f;
    for (int j = lane; j < W2DIM; j += 32) p += sh_x[bl][j] * w2[j];
    float s = wred(p);
    if (lane == 0) out[bg * T_ITM + t] = s + b2_w[idx];
  }
}

extern "C" void kernel_launch(void* const* d_in, const int* in_sizes, int n_in,
                              void* d_out, int out_size, void* d_ws, size_t ws_size,
                              hipStream_t stream) {
  (void)in_sizes; (void)n_in; (void)out_size; (void)d_ws; (void)ws_size;
  const int*   seq_var    = (const int*)d_in[0];
  const int*   user_var   = (const int*)d_in[1];
  const int*   item_var   = (const int*)d_in[2];
  const float* item_emb_w = (const float*)d_in[3];
  const float* user_emb_w = (const float*)d_in[4];
  const float* conv_v_w   = (const float*)d_in[5];
  const float* conv_v_b   = (const float*)d_in[6];
  const float* conv_h_w   = (const float*)d_in[7];
  const float* conv_h_b   = (const float*)d_in[8];
  const float* conv_u_w   = (const float*)d_in[9];
  const float* conv_u_b   = (const float*)d_in[10];
  const float* conv_u1_w  = (const float*)d_in[11];
  const float* conv_u1_b  = (const float*)d_in[12];
  const float* W2_w       = (const float*)d_in[13];
  const float* b2_w       = (const float*)d_in[14];
  const float* CW1        = (const float*)d_in[15];
  const float* CW2        = (const float*)d_in[16];
  const float* CW00       = (const float*)d_in[17];
  const float* CW11       = (const float*)d_in[18];
  float* out = (float*)d_out;

  dim3 grid(B_TOT / BPB), block(NTHREADS);
  hipLaunchKernelGGL(caser_fused_kernel, grid, block, 0, stream,
                     seq_var, user_var, item_var, item_emb_w, user_emb_w,
                     conv_v_w, conv_v_b, conv_h_w, conv_h_b,
                     conv_u_w, conv_u_b, conv_u1_w, conv_u1_b,
                     W2_w, b2_w, CW1, CW2, CW00, CW11, out);
}
